// PLE_Layer_38036230374041
// MI455X (gfx1250) — compile-verified, matrix operand order checked by reading, not run on hardware
//
#include <hip/hip_runtime.h>

// PLE layer for MI455X (gfx1250, wave32, WMMA).
// B=4, S=2048 -> 8192 tokens; H=512; E=8 experts per group.
//
// Pipeline:
//   repack_weights : fp32 W[k,h,e] -> bf16 WMMA-B fragments in d_ws
//   ln_gate_pack   : LayerNorm -> bf16 WMMA-A fragments + softmax gate probs
//   expert_combine : fused 24-way bf16 WMMA GEMM + bias + relu + gated
//                    cross-stream combine in LDS -> 3 fp32 outputs

#define HH   512
#define EE   8
#define NTOK 8192

typedef __attribute__((ext_vector_type(16))) __bf16 v16bf;
typedef __attribute__((ext_vector_type(8)))  float  v8f;

// ---- workspace layout (bytes) ----
#define WS_APACK 0u                 // 3 * 8 MB  (A fragments, bf16)
#define WS_BPACK 25165824u          // 3 * 4 MB  (B fragments, bf16)
#define WS_GSH   37748736u          // 8192*24 fp32
#define WS_GT1   38535168u          // 8192*16 fp32
#define WS_GT2   39059456u          // 8192*16 fp32
// total: 39583744 bytes (~37.8 MB)

__device__ __forceinline__ unsigned f2bf(float f) {
  union { float f; unsigned u; } c; c.f = f;
  unsigned u = c.u;
  u += 0x7fffu + ((u >> 16) & 1u);   // round-to-nearest-even
  return u >> 16;
}

__device__ __forceinline__ void lds_fadd(float* p, float v) {
  // returnless LDS float add -> ds_add_f32
  __hip_atomic_fetch_add(p, v, __ATOMIC_RELAXED, __HIP_MEMORY_SCOPE_WORKGROUP);
}

// ---------------------------------------------------------------------------
// Kernel 0: repack fp32 W[k][h][e] into per-(stream,expert) bf16 B fragments.
// Fragment = 16(N) x 32(K) bf16 = 1KB, lane-contiguous 32B per lane so the
// GEMM loads it with two b128 loads directly into WMMA B layout:
//   lane l: n = l%16, g = l/16 ; VGPR v<4: k = 8g+2v+{0,1}; v>=4: k = 16+8g+2(v-4)+{0,1}
// ---------------------------------------------------------------------------
__global__ __launch_bounds__(256) void repack_weights(
    const float* __restrict__ Wsh, const float* __restrict__ Wt1,
    const float* __restrict__ Wt2, unsigned char* __restrict__ bpack) {
  int frag = blockIdx.x * 8 + (threadIdx.x >> 5);   // 12288 frags total
  int lane = threadIdx.x & 31;
  int s    = frag >> 12;          // 4096 frags per stream
  int rem  = frag & 4095;         // = (e*32 + ntg)*16 + kc
  int e    = rem >> 9;
  int rem2 = rem & 511;
  int kc   = rem2 & 15;
  int n    = ((rem2 >> 4) << 4) + (lane & 15);
  int g    = lane >> 4;
  const float* W = (s == 0) ? Wsh : (s == 1) ? Wt1 : Wt2;
  unsigned dws[8];
#pragma unroll
  for (int v = 0; v < 8; ++v) {
    int k0 = kc * 32 + ((v < 4) ? (8 * g + 2 * v) : (16 + 8 * g + 2 * (v - 4)));
    float w0 = W[(size_t)(k0 * HH + n) * EE + e];
    float w1 = W[(size_t)((k0 + 1) * HH + n) * EE + e];
    dws[v] = f2bf(w0) | (f2bf(w1) << 16);
  }
  unsigned char* dst = bpack + (size_t)s * 4194304u + (size_t)rem * 1024u + lane * 32;
  ((uint4*)dst)[0] = make_uint4(dws[0], dws[1], dws[2], dws[3]);
  ((uint4*)dst)[1] = make_uint4(dws[4], dws[5], dws[6], dws[7]);
}

// ---------------------------------------------------------------------------
// Kernel 1: per (stream, token): LayerNorm, write bf16 A fragments, gate
// logits + softmax probs. One block (256 thr) per token per stream.
// ---------------------------------------------------------------------------
__global__ __launch_bounds__(256) void ln_gate_pack(
    const float* __restrict__ x0, const float* __restrict__ x1,
    const float* __restrict__ x2,
    const float* __restrict__ g0, const float* __restrict__ b0,
    const float* __restrict__ g1, const float* __restrict__ b1,
    const float* __restrict__ g2, const float* __restrict__ b2,
    const float* __restrict__ Wg0, const float* __restrict__ bg0,
    const float* __restrict__ Wg1, const float* __restrict__ bg1,
    const float* __restrict__ Wg2, const float* __restrict__ bg2,
    unsigned char* __restrict__ apack,
    float* __restrict__ gsh, float* __restrict__ gt1, float* __restrict__ gt2) {
  __shared__ float xn[HH];
  __shared__ float rs[256], rq[256];
  __shared__ float logits[24];
  int m = blockIdx.x, s = blockIdx.y, t = threadIdx.x;
  const float* x  = (s == 0) ? x0  : (s == 1) ? x1  : x2;
  const float* lg = (s == 0) ? g0  : (s == 1) ? g1  : g2;
  const float* lb = (s == 0) ? b0  : (s == 1) ? b1  : b2;
  const float* Wg = (s == 0) ? Wg0 : (s == 1) ? Wg1 : Wg2;
  const float* bg = (s == 0) ? bg0 : (s == 1) ? bg1 : bg2;
  int NL = (s == 0) ? 24 : 16;

  float2 xv = ((const float2*)(x + (size_t)m * HH))[t];
  rs[t] = xv.x + xv.y;
  rq[t] = xv.x * xv.x + xv.y * xv.y;
  __syncthreads();
  for (int st = 128; st > 0; st >>= 1) {
    if (t < st) { rs[t] += rs[t + st]; rq[t] += rq[t + st]; }
    __syncthreads();
  }
  float mean = rs[0] * (1.f / HH);
  float var  = rq[0] * (1.f / HH) - mean * mean;
  float rstd = rsqrtf(var + 1e-5f);
  float y0 = (xv.x - mean) * rstd * lg[2 * t]     + lb[2 * t];
  float y1 = (xv.y - mean) * rstd * lg[2 * t + 1] + lb[2 * t + 1];
  xn[2 * t] = y0; xn[2 * t + 1] = y1;

  // store bf16 pair into WMMA-A fragment layout (one dword per thread)
  int kk0 = 2 * t, kc = kk0 >> 5, kk = kk0 & 31;
  int g, v;
  if (kk < 16) { g = kk >> 3; v = (kk & 7) >> 1; }
  else         { int r2 = kk - 16; g = r2 >> 3; v = 4 + ((r2 & 7) >> 1); }
  int lane = g * 16 + (m & 15);
  unsigned dw = f2bf(y0) | (f2bf(y1) << 16);
  *(unsigned*)(apack + (size_t)s * 8388608u +
               (size_t)(((m >> 4) * 16 + kc)) * 1024u + lane * 32 + v * 4) = dw;
  __syncthreads();

  // gate logits: wave w computes logits {w, w+8, w+16}
  int wv = t >> 5, ln = t & 31;
#pragma unroll
  for (int jj = 0; jj < 3; ++jj) {
    int j = wv + jj * 8;
    if (j < NL) {
      float p = 0.f;
      for (int k = ln; k < HH; k += 32) p += xn[k] * Wg[k * NL + j];
      for (int off = 16; off > 0; off >>= 1) p += __shfl_down(p, off);
      if (ln == 0) logits[j] = p + bg[j];
    }
  }
  __syncthreads();
  if (t == 0) {
    float mx = -1e30f;
    for (int j = 0; j < NL; ++j) mx = fmaxf(mx, logits[j]);
    float sum = 0.f, pr[24];
    for (int j = 0; j < NL; ++j) { pr[j] = __expf(logits[j] - mx); sum += pr[j]; }
    float inv = 1.f / sum;
    float* dst = (s == 0) ? gsh + (size_t)m * 24
               : (s == 1) ? gt1 + (size_t)m * 16 : gt2 + (size_t)m * 16;
    for (int j = 0; j < NL; ++j) dst[j] = pr[j] * inv;
  }
}

// ---------------------------------------------------------------------------
// Kernel 2: fused expert GEMMs (bf16 WMMA) + bias + relu + gated combine.
// Workgroup tile: [64 tokens x 64 h]; 8 waves, wave w = expert w, streams
// processed sequentially; partial sums reduced across waves via ds_add_f32.
// __launch_bounds__(256, 4): cap at 256 VGPR/wave. Epilogue processes one
// m-tile at a time (<=24 gate coeffs live) to stay under the cap (no spills).
// ---------------------------------------------------------------------------
__global__ __launch_bounds__(256, 4) void expert_combine(
    const unsigned char* __restrict__ apack,
    const unsigned char* __restrict__ bpack,
    const float* __restrict__ gsh, const float* __restrict__ gt1,
    const float* __restrict__ gt2,
    const float* __restrict__ bsh, const float* __restrict__ bt1,
    const float* __restrict__ bt2,
    float* __restrict__ out) {
  __shared__ float acc_sh[64 * 64];
  __shared__ float acc_t1[64 * 64];
  __shared__ float acc_t2[64 * 64];
  __shared__ float gbuf[3584];   // [64][24] gsh | [64][16] gt1 | [64][16] gt2
  int cb = blockIdx.x, rb = blockIdx.y, t = threadIdx.x;
  int wave = t >> 5, lane = t & 31;
  int e = wave;
  int nl = lane & 15, gsel = lane >> 4;
  int tok0 = rb * 64;

  for (int i = t; i < 4096; i += 256) { acc_sh[i] = 0.f; acc_t1[i] = 0.f; acc_t2[i] = 0.f; }
  for (int i = t; i < 1536; i += 256) gbuf[i]        = gsh[(size_t)tok0 * 24 + i];
  for (int i = t; i < 1024; i += 256) gbuf[1536 + i] = gt1[(size_t)tok0 * 16 + i];
  for (int i = t; i < 1024; i += 256) gbuf[2560 + i] = gt2[(size_t)tok0 * 16 + i];
  __syncthreads();

  for (int s = 0; s < 3; ++s) {
    const unsigned char* Ab = apack + (size_t)s * 8388608u +
                              (size_t)(rb * 4) * 16u * 1024u + lane * 32;
    const unsigned char* Bb = bpack + (size_t)s * 4194304u +
                              (size_t)((e * 32 + cb * 4) * 16) * 1024u + lane * 32;
    v8f acc[4][4];
#pragma unroll
    for (int mt = 0; mt < 4; ++mt)
#pragma unroll
      for (int nt = 0; nt < 4; ++nt)
#pragma unroll
        for (int r = 0; r < 8; ++r) acc[mt][nt][r] = 0.f;

    for (int kc = 0; kc < 16; ++kc) {
      // one load clause: 4 B-frags + 4 A-frags (16 x b128), then 16 WMMAs
      v16bf bfr[4], afr[4];
#pragma unroll
      for (int nt = 0; nt < 4; ++nt)
        bfr[nt] = *(const v16bf*)(Bb + (size_t)(nt * 16 + kc) * 1024u);
#pragma unroll
      for (int mt = 0; mt < 4; ++mt)
        afr[mt] = *(const v16bf*)(Ab + (size_t)(mt * 16 + kc) * 1024u);
#pragma unroll
      for (int mt = 0; mt < 4; ++mt)
#pragma unroll
        for (int nt = 0; nt < 4; ++nt)
          acc[mt][nt] = __builtin_amdgcn_wmma_f32_16x16x32_bf16(
              false, afr[mt], false, bfr[nt], (short)0, acc[mt][nt], false, false);
    }

    const float* bias = (s == 0) ? bsh : (s == 1) ? bt1 : bt2;
    float bv[4];
#pragma unroll
    for (int nt = 0; nt < 4; ++nt) bv[nt] = bias[(cb * 64 + nt * 16 + nl) * EE + e];

    // Epilogue: one m-tile at a time to bound live gate coefficients.
#pragma unroll
    for (int mt = 0; mt < 4; ++mt) {
      float ga[8], gb[8], gc[8];   // coeffs: ->out_sh, ->out_t1, ->out_t2
#pragma unroll
      for (int r = 0; r < 8; ++r) {
        int tl = mt * 16 + gsel * 8 + r;
        if (s == 0) {              // shared expert feeds all three outputs
          ga[r] = gbuf[tl * 24 + 8 + e];
          gb[r] = gbuf[1536 + tl * 16 + 8 + e];
          gc[r] = gbuf[2560 + tl * 16 + 8 + e];
        } else if (s == 1) {       // task1 expert -> out_sh slice0, out_t1 slice0
          ga[r] = gbuf[tl * 24 + e];
          gb[r] = gbuf[1536 + tl * 16 + e];
          gc[r] = 0.f;
        } else {                   // task2 expert -> out_sh slice2, out_t2 slice0
          ga[r] = gbuf[tl * 24 + 16 + e];
          gb[r] = 0.f;
          gc[r] = gbuf[2560 + tl * 16 + e];
        }
      }
#pragma unroll
      for (int nt = 0; nt < 4; ++nt)
#pragma unroll
        for (int r = 0; r < 8; ++r) {
          float vv = fmaxf(acc[mt][nt][r] + bv[nt], 0.f);   // relu(x*W + b)
          int oi = (mt * 16 + gsel * 8 + r) * 64 + nt * 16 + nl;
          lds_fadd(&acc_sh[oi], ga[r] * vv);
          if (s != 2) lds_fadd(&acc_t1[oi], gb[r] * vv);
          if (s != 1) lds_fadd(&acc_t2[oi], gc[r] * vv);
        }
    }
  }
  __syncthreads();

  int r0 = t >> 4, c4 = (t & 15) * 4;
#pragma unroll
  for (int o = 0; o < 3; ++o) {
    float* dst = out + (size_t)o * 4194304u;
    const float* src = (o == 0) ? acc_sh : (o == 1) ? acc_t1 : acc_t2;
    for (int rr = r0; rr < 64; rr += 16) {
      float4 vv = *(const float4*)&src[rr * 64 + c4];
      *(float4*)(dst + (size_t)(tok0 + rr) * HH + cb * 64 + c4) = vv;
    }
  }
}

// ---------------------------------------------------------------------------
extern "C" void kernel_launch(void* const* d_in, const int* in_sizes, int n_in,
                              void* d_out, int out_size, void* d_ws, size_t ws_size,
                              hipStream_t stream) {
  (void)in_sizes; (void)n_in; (void)out_size; (void)ws_size;
  const float* x_sh = (const float*)d_in[0];
  const float* x_t1 = (const float*)d_in[1];
  const float* x_t2 = (const float*)d_in[2];
  const float* ln0g = (const float*)d_in[3];
  const float* ln0b = (const float*)d_in[4];
  const float* ln1g = (const float*)d_in[5];
  const float* ln1b = (const float*)d_in[6];
  const float* ln2g = (const float*)d_in[7];
  const float* ln2b = (const float*)d_in[8];
  const float* W_sh = (const float*)d_in[9];
  const float* b_sh = (const float*)d_in[10];
  const float* W_t1 = (const float*)d_in[11];
  const float* b_t1 = (const float*)d_in[12];
  const float* W_t2 = (const float*)d_in[13];
  const float* b_t2 = (const float*)d_in[14];
  const float* Wg_sh = (const float*)d_in[15];
  const float* bg_sh = (const float*)d_in[16];
  const float* Wg_t1 = (const float*)d_in[17];
  const float* bg_t1 = (const float*)d_in[18];
  const float* Wg_t2 = (const float*)d_in[19];
  const float* bg_t2 = (const float*)d_in[20];

  unsigned char* ws = (unsigned char*)d_ws;
  unsigned char* apack = ws + WS_APACK;
  unsigned char* bpack = ws + WS_BPACK;
  float* gsh = (float*)(ws + WS_GSH);
  float* gt1 = (float*)(ws + WS_GT1);
  float* gt2 = (float*)(ws + WS_GT2);

  repack_weights<<<dim3(1536), dim3(256), 0, stream>>>(W_sh, W_t1, W_t2, bpack);
  ln_gate_pack<<<dim3(NTOK, 3), dim3(256), 0, stream>>>(
      x_sh, x_t1, x_t2, ln0g, ln0b, ln1g, ln1b, ln2g, ln2b,
      Wg_sh, bg_sh, Wg_t1, bg_t1, Wg_t2, bg_t2, apack, gsh, gt1, gt2);
  expert_combine<<<dim3(8, 128), dim3(256), 0, stream>>>(
      apack, bpack, gsh, gt1, gt2, b_sh, b_t1, b_t2, (float*)d_out);
}
